// DeltaNetBlock_27230092657575
// MI455X (gfx1250) — compile-verified, matrix operand order checked by reading, not run on hardware
//
#include <hip/hip_runtime.h>
#include <hip/hip_bf16.h>
#include <stdint.h>

// ---------------------------------------------------------------------------
// Types for CDNA5 WMMA (wave32): v_wmma_f32_16x16x32_bf16
// ---------------------------------------------------------------------------
typedef __attribute__((ext_vector_type(16))) __bf16 v16bf;
typedef __attribute__((ext_vector_type(8)))  float  v8f;

union AFrag { v16bf v; uint4 u[2]; };

static __device__ __forceinline__ __bf16 f2bf(float f) {
    union { float f; uint32_t u; } in; in.f = f;
    uint32_t u = in.u;
    uint32_t r = (u + 0x7FFFu + ((u >> 16) & 1u)) >> 16;   // RNE
    union { unsigned short s; __bf16 b; } out; out.s = (unsigned short)r;
    return out.b;
}

static __device__ __forceinline__ float sigmoidf_(float x) {
    return 1.0f / (1.0f + __expf(-x));
}
static __device__ __forceinline__ float siluf_(float x) {
    return x * sigmoidf_(x);
}

// ---------------------------------------------------------------------------
// Problem constants (reference: B=2, T=2048, D=1024, F=4096, H=8, DH=128)
// ---------------------------------------------------------------------------
#define BB 2
#define TT 2048
#define DD 1024
#define FF 4096
#define HH 8
#define DHD 128
#define ROWS (BB*TT)       // 4096

// ---------------------------------------------------------------------------
// float -> bf16 conversion (weights, done once per launch)
// ---------------------------------------------------------------------------
__global__ void cvt_bf16_kernel(const float* __restrict__ in,
                                __bf16* __restrict__ out, int n) {
    int i = blockIdx.x * 256 + threadIdx.x;
    if (i < n) out[i] = f2bf(in[i]);
}

// ---------------------------------------------------------------------------
// RMSNorm over last dim (D=1024). One block (256 thr) per row. fp32 out.
// ---------------------------------------------------------------------------
__global__ void rmsnorm_kernel(const float* __restrict__ x,
                               float* __restrict__ y) {
    const int row = blockIdx.x;
    const int tid = threadIdx.x;
    __shared__ float red[256];
    float s = 0.f;
    for (int i = tid; i < DD; i += 256) {
        float v = x[(size_t)row * DD + i];
        s += v * v;
    }
    red[tid] = s;
    __syncthreads();
    for (int off = 128; off > 0; off >>= 1) {
        if (tid < off) red[tid] += red[tid + off];
        __syncthreads();
    }
    const float scale = rsqrtf(red[0] * (1.0f / (float)DD) + 1e-6f);
    for (int i = tid; i < DD; i += 256)
        y[(size_t)row * DD + i] = x[(size_t)row * DD + i] * scale;
}

// ---------------------------------------------------------------------------
// Per-head L2 norm in place: rows of 128 (q/k after silu). Block = 128 thr.
// ---------------------------------------------------------------------------
__global__ void l2norm_kernel(float* __restrict__ p) {
    const int row = blockIdx.x;          // (b*T + t)*H + h
    const int tid = threadIdx.x;         // 0..127
    __shared__ float red[128];
    const float v = p[(size_t)row * DHD + tid];
    red[tid] = v * v;
    __syncthreads();
    for (int off = 64; off > 0; off >>= 1) {
        if (tid < off) red[tid] += red[tid + off];
        __syncthreads();
    }
    const float scale = rsqrtf(red[0] + 1e-6f);
    p[(size_t)row * DHD + tid] = v * scale;
}

// ---------------------------------------------------------------------------
// beta / alpha gates: sigmoid(xn @ wb), sigmoid(xn @ wg); wb,wg: [D, H]
// One block (256 thr) per row. thread -> (output col 0..15, segment 0..15)
// ---------------------------------------------------------------------------
__global__ void gates_kernel(const float* __restrict__ xn,
                             const float* __restrict__ wb,
                             const float* __restrict__ wg,
                             float* __restrict__ beta,
                             float* __restrict__ alpha) {
    const int row = blockIdx.x;
    const int tid = threadIdx.x;
    const int col = tid & 15;            // 0..7 beta, 8..15 alpha
    const int seg = tid >> 4;            // 16 segments of 64
    const float* w = (col < 8) ? wb : wg;
    const int c = col & 7;
    float s = 0.f;
    const int i0 = seg * 64;
    for (int i = i0; i < i0 + 64; ++i)
        s += xn[(size_t)row * DD + i] * w[(size_t)i * HH + c];
    __shared__ float red[256];
    red[tid] = s;
    __syncthreads();
    if (tid < 16) {
        float t = 0.f;
        for (int g2 = 0; g2 < 16; ++g2) t += red[g2 * 16 + tid];
        const float sg = sigmoidf_(t);
        if (tid < 8) beta[(size_t)row * HH + tid] = sg;
        else         alpha[(size_t)row * HH + (tid - 8)] = sg;
    }
}

// ---------------------------------------------------------------------------
// GEMM: C[M,N] (+epilogue) = A[M,K] (fp32, converted to bf16 in LDS staging)
//                            x B[K,N] (bf16, pre-converted weights)
// Block tile 128x128, BK=64 (2 WMMA k-steps per stage), 256 threads = 8 waves
// (4 over M x 2 over N), each wave computes 2x4 tiles of 16x16 with
// v_wmma_f32_16x16x32_bf16. launch_bounds(256,1): big VGPR budget, no spills.
// ---------------------------------------------------------------------------
enum { EPI_NONE = 0, EPI_SILU = 1, EPI_ADDRES = 2, EPI_MUL = 3 };

template <int EPI>
__global__ void __launch_bounds__(256, 1)
gemm_bf16_kernel(const float* __restrict__ A,
                 const __bf16* __restrict__ B,
                 float* __restrict__ C,
                 const float* __restrict__ aux,
                 int M, int N, int K) {
    constexpr int BM = 128, BN = 128, BK = 64;
    constexpr int LDT = 72;  // padded bf16 stride (144B, 16B-aligned rows)

    __shared__ alignas(16) __bf16 As[BM * LDT];   // A tile, row-major [m][k]
    __shared__ alignas(16) __bf16 Bs[BN * LDT];   // B tile transposed [n][k]

    const int tid  = threadIdx.x;
    const int lane = tid & 31;
    const int wave = tid >> 5;
    const int wm = (wave & 3) * 32;   // wave row offset inside block tile
    const int wn = (wave >> 2) * 64;  // wave col offset inside block tile
    const int lrow = lane & 15;
    const int kh   = (lane >> 4) * 8; // K half select per ISA A-layout

    const int n0 = blockIdx.x * BN;
    const int m0 = blockIdx.y * BM;

    v8f acc[2][4];
    const v8f vzero = {0.f, 0.f, 0.f, 0.f, 0.f, 0.f, 0.f, 0.f};
#pragma unroll
    for (int i = 0; i < 2; ++i)
#pragma unroll
        for (int j = 0; j < 4; ++j) acc[i][j] = vzero;

    for (int k0 = 0; k0 < K; k0 += BK) {
        // ---- stage A (fp32 -> bf16): 128x64, 32 elems per thread ----
#pragma unroll
        for (int i = tid; i < BM * BK; i += 256) {
            const int r = i >> 6;        // /BK
            const int c = i & 63;
            As[r * LDT + c] = f2bf(A[(size_t)(m0 + r) * K + k0 + c]);
        }
        // ---- stage B transposed: Bs[n][k] = B[k0+k][n0+n] ----
#pragma unroll
        for (int i = tid; i < BK * BN; i += 256) {
            const int n = i & 127;
            const int k = i >> 7;
            Bs[n * LDT + k] = B[(size_t)(k0 + k) * N + n0 + n];
        }
        // prefetch next K tile into L2 (global_prefetch_b8)
        if (k0 + BK < K) {
            __builtin_prefetch(A + (size_t)(m0 + (tid >> 1)) * K + k0 + BK + (tid & 1) * 32, 0, 1);
            __builtin_prefetch(B + (size_t)(k0 + BK + (tid >> 3)) * N + n0 + (tid & 7) * 16, 0, 1);
        }
        __syncthreads();

        // ---- 2 WMMA k-steps of 32 per staged tile ----
#pragma unroll
        for (int kk = 0; kk < BK; kk += 32) {
            AFrag af[2], bf[4];
#pragma unroll
            for (int i = 0; i < 2; ++i) {
                const __bf16* p = As + (wm + i * 16 + lrow) * LDT + kk + kh;
                af[i].u[0] = *(const uint4*)(p);        // K[kh .. kh+8)
                af[i].u[1] = *(const uint4*)(p + 16);   // K[kh+16 .. kh+24)
            }
#pragma unroll
            for (int j = 0; j < 4; ++j) {
                const __bf16* p = Bs + (wn + j * 16 + lrow) * LDT + kk + kh;
                bf[j].u[0] = *(const uint4*)(p);
                bf[j].u[1] = *(const uint4*)(p + 16);
            }
#pragma unroll
            for (int i = 0; i < 2; ++i)
#pragma unroll
                for (int j = 0; j < 4; ++j)
                    acc[i][j] = __builtin_amdgcn_wmma_f32_16x16x32_bf16(
                        false, af[i].v, false, bf[j].v,
                        (short)0, acc[i][j], false, false);
        }
        __syncthreads();
    }

    // ---- epilogue / store; C/D layout: vgpr v -> M=v (+8 for lanes 16..31),
    //      N = lane&15 ----
    const int mo = (lane >> 4) * 8;
#pragma unroll
    for (int i = 0; i < 2; ++i) {
#pragma unroll
        for (int j = 0; j < 4; ++j) {
            const int tm = m0 + wm + i * 16;
            const int tn = n0 + wn + j * 16 + (lane & 15);
#pragma unroll
            for (int v2 = 0; v2 < 8; ++v2) {
                const int m = tm + v2 + mo;
                const size_t idx = (size_t)m * N + tn;
                float val = acc[i][j][v2];
                if (EPI == EPI_SILU)        val = siluf_(val);
                else if (EPI == EPI_ADDRES) val = val + aux[idx];
                else if (EPI == EPI_MUL)    val = val * aux[idx];
                C[idx] = val;
            }
        }
    }
}

// ---------------------------------------------------------------------------
// Gated DeltaNet recurrence. One block per (b,h); 256 threads.
// State column S[:,col] is register-resident, K-dim split across 2 halves
// (64 fp32 regs per lane). LDS only for k/q broadcast and partial sums.
// ---------------------------------------------------------------------------
__global__ void __launch_bounds__(256, 1)
scan_kernel(const float* __restrict__ qg,
            const float* __restrict__ kg,
            const float* __restrict__ vg,
            const float* __restrict__ betag,
            const float* __restrict__ alphag,
            const float* __restrict__ S0,
            float* __restrict__ obuf,
            float* __restrict__ Sout) {
    const int bh  = blockIdx.x;          // b*H + h
    const int b   = bh >> 3;
    const int h   = bh & 7;
    const int tid = threadIdx.x;         // 256
    const int col = tid & 127;           // state column (v index)
    const int half = tid >> 7;           // k-range half
    const int rb = half * 64;

    __shared__ float ks[128];
    __shared__ float qs[128];
    __shared__ float red[256];

    float S[64];
#pragma unroll
    for (int i = 0; i < 64; ++i)
        S[i] = S0[((size_t)bh * DHD + rb + i) * DHD + col];

    for (int t = 0; t < TT; ++t) {
        const size_t base = ((size_t)(b * TT + t)) * (HH * DHD) + h * DHD;
        if (half == 0) {
            ks[col] = kg[base + col];
            qs[col] = qg[base + col];
        }
        const float vt = vg[base + col];
        const float bt = betag[(size_t)(b * TT + t) * HH + h];
        const float at = alphag[(size_t)(b * TT + t) * HH + h];
        __syncthreads();

        // pred_v = alpha * sum_k k[k] * S[k][v]   (decay folded out of sum)
        float p = 0.f;
#pragma unroll
        for (int i = 0; i < 64; ++i) p += ks[rb + i] * S[i];
        red[tid] = p;
        __syncthreads();
        const float pred = at * (red[col] + red[128 + col]);
        const float dv = bt * (vt - pred);

        // S = alpha*S + k (x) dv ;  o_v = sum_k q[k] * S_new[k][v]
        float o = 0.f;
#pragma unroll
        for (int i = 0; i < 64; ++i) {
            const float sn = at * S[i] + ks[rb + i] * dv;
            S[i] = sn;
            o += qs[rb + i] * sn;
        }
        red[tid] = o;
        __syncthreads();
        if (half == 0) obuf[base + col] = red[col] + red[128 + col];
        __syncthreads();   // protect ks/qs/red before next step overwrites
    }

#pragma unroll
    for (int i = 0; i < 64; ++i)
        Sout[((size_t)bh * DHD + rb + i) * DHD + col] = S[i];
}

// ---------------------------------------------------------------------------
// Host-side orchestration
// ---------------------------------------------------------------------------
extern "C" void kernel_launch(void* const* d_in, const int* in_sizes, int n_in,
                              void* d_out, int out_size, void* d_ws, size_t ws_size,
                              hipStream_t stream) {
    (void)in_sizes; (void)n_in; (void)out_size; (void)ws_size;

    const float* x      = (const float*)d_in[0];
    const float* S0     = (const float*)d_in[1];
    const float* wq     = (const float*)d_in[2];
    const float* wk     = (const float*)d_in[3];
    const float* wv     = (const float*)d_in[4];
    const float* wb     = (const float*)d_in[5];
    const float* wg     = (const float*)d_in[6];
    const float* wo     = (const float*)d_in[7];
    const float* wgate  = (const float*)d_in[8];
    const float* wup    = (const float*)d_in[9];
    const float* wdown  = (const float*)d_in[10];

    float* out = (float*)d_out;                 // [B,T,D] then [B,H,DH,DH]
    float* Sout = out + (size_t)BB * TT * DD;   // offset 4,194,304

    char* ws = (char*)d_ws;
    const size_t MB = 1024ull * 1024ull;
    __bf16* wq_bf    = (__bf16*)(ws + 0 * MB);    // 2 MB
    __bf16* wk_bf    = (__bf16*)(ws + 2 * MB);    // 2 MB
    __bf16* wv_bf    = (__bf16*)(ws + 4 * MB);    // 2 MB
    __bf16* wo_bf    = (__bf16*)(ws + 6 * MB);    // 2 MB
    __bf16* wgate_bf = (__bf16*)(ws + 8 * MB);    // 8 MB
    __bf16* wup_bf   = (__bf16*)(ws + 16 * MB);   // 8 MB
    __bf16* wdown_bf = (__bf16*)(ws + 24 * MB);   // 8 MB
    float* xn    = (float*)(ws + 32 * MB);        // 16 MB
    float* qb    = (float*)(ws + 48 * MB);        // 16 MB
    float* kb    = (float*)(ws + 64 * MB);        // 16 MB
    float* vb    = (float*)(ws + 80 * MB);        // 16 MB
    float* betab = (float*)(ws + 96 * MB);        // 128 KB
    float* alphb = (float*)(ws + 96 * MB + 256 * 1024); // 128 KB
    float* obuf  = (float*)(ws + 97 * MB);        // 16 MB
    float* h1    = (float*)(ws + 113 * MB);       // 16 MB
    float* hn    = (float*)(ws + 129 * MB);       // 16 MB
    float* gbuf  = (float*)(ws + 145 * MB);       // 64 MB  (g, then g*u)

    const int nDW = DD * HH * DHD;  // 1,048,576 (wq/wk/wv/wo)
    const int nDF = DD * FF;        // 4,194,304 (w_gate/w_up/w_down)

    // 1) weights -> bf16
    cvt_bf16_kernel<<<(nDW + 255) / 256, 256, 0, stream>>>(wq, wq_bf, nDW);
    cvt_bf16_kernel<<<(nDW + 255) / 256, 256, 0, stream>>>(wk, wk_bf, nDW);
    cvt_bf16_kernel<<<(nDW + 255) / 256, 256, 0, stream>>>(wv, wv_bf, nDW);
    cvt_bf16_kernel<<<(nDW + 255) / 256, 256, 0, stream>>>(wo, wo_bf, nDW);
    cvt_bf16_kernel<<<(nDF + 255) / 256, 256, 0, stream>>>(wgate, wgate_bf, nDF);
    cvt_bf16_kernel<<<(nDF + 255) / 256, 256, 0, stream>>>(wup, wup_bf, nDF);
    cvt_bf16_kernel<<<(nDF + 255) / 256, 256, 0, stream>>>(wdown, wdown_bf, nDF);

    // 2) xn = rms_norm(x)
    rmsnorm_kernel<<<ROWS, 256, 0, stream>>>(x, xn);

    // 3) q/k/v = silu(xn @ W)   (WMMA bf16, silu fused)
    dim3 gDW(HH * DHD / 128, ROWS / 128);  // (8, 32)
    gemm_bf16_kernel<EPI_SILU><<<gDW, 256, 0, stream>>>(xn, wq_bf, qb, nullptr, ROWS, HH * DHD, DD);
    gemm_bf16_kernel<EPI_SILU><<<gDW, 256, 0, stream>>>(xn, wk_bf, kb, nullptr, ROWS, HH * DHD, DD);
    gemm_bf16_kernel<EPI_SILU><<<gDW, 256, 0, stream>>>(xn, wv_bf, vb, nullptr, ROWS, HH * DHD, DD);

    // 4) beta/alpha gates
    gates_kernel<<<ROWS, 256, 0, stream>>>(xn, wb, wg, betab, alphb);

    // 5) per-head l2norm on q,k (in place)
    l2norm_kernel<<<ROWS * HH, 128, 0, stream>>>(qb);
    l2norm_kernel<<<ROWS * HH, 128, 0, stream>>>(kb);

    // 6) gated delta-rule scan -> obuf, S_final
    scan_kernel<<<BB * HH, 256, 0, stream>>>(qb, kb, vb, betab, alphb, S0, obuf, Sout);

    // 7) h1 = x + obuf @ wo     (residual fused)
    gemm_bf16_kernel<EPI_ADDRES><<<gDW, 256, 0, stream>>>(obuf, wo_bf, h1, x, ROWS, DD, HH * DHD);

    // 8) hn = rms_norm(h1)
    rmsnorm_kernel<<<ROWS, 256, 0, stream>>>(h1, hn);

    // 9) g = silu(hn @ w_gate); g = g * (hn @ w_up); out = h1 + g @ w_down
    dim3 gDF(FF / 128, ROWS / 128);        // (32, 32)
    gemm_bf16_kernel<EPI_SILU><<<gDF, 256, 0, stream>>>(hn, wgate_bf, gbuf, nullptr, ROWS, FF, DD);
    gemm_bf16_kernel<EPI_MUL><<<gDF, 256, 0, stream>>>(hn, wup_bf, gbuf, gbuf, ROWS, FF, DD);
    gemm_bf16_kernel<EPI_ADDRES><<<gDW, 256, 0, stream>>>(gbuf, wdown_bf, out, h1, ROWS, DD, FF);
}
